// DHGCF_11269994184844
// MI455X (gfx1250) — matrix operand order repaired and checked by reading, compiled, not run on hardware
//
#include <hip/hip_runtime.h>
#include <hip/hip_bf16.h>
#include <math.h>

// MI455X (gfx1250, wave32). Workload is HBM-bound (~2.2 GB traffic -> ~95us
// at 23.3 TB/s). Dense 64x64 layer GEMMs use v_wmma_f32_16x16x32_f16 with
// f32 accumulate; W is staged into LDS by the Tensor Data Mover
// (tensor_load_to_lds + s_wait_tensorcnt) and re-laid-out into the exact
// WMMA B-fragment order so each fragment is one 32B LDS read.

typedef __attribute__((ext_vector_type(16))) _Float16     v16h;
typedef __attribute__((ext_vector_type(8)))  float        v8f;
typedef __attribute__((ext_vector_type(4)))  unsigned int u32x4;
typedef __attribute__((ext_vector_type(8)))  int          i32x8;
typedef __attribute__((ext_vector_type(4)))  int          i32x4;

#define KEEP24 15099494u   // floor(0.9 * 2^24)  (edge dropout)
#define KEEP8  230u        // floor(0.9 * 256)   (message dropout, per-byte)
#define INV_KEEP (1.0f / 0.9f)

__device__ __forceinline__ unsigned hash_u32(unsigned x) {
    x ^= x >> 16; x *= 0x7feb352du;
    x ^= x >> 15; x *= 0x846ca68bu;
    x ^= x >> 16;
    return x;
}

// ---------------------------------------------------------------- edge dropout
__global__ void edge_drop_kernel(const float* __restrict__ vals,
                                 float* __restrict__ a_vals, int E) {
    int e = blockIdx.x * blockDim.x + threadIdx.x;
    if (e >= E) return;
    unsigned h = hash_u32(0xA5A5F00Du ^ (unsigned)e);
    a_vals[e] = ((h & 0xFFFFFFu) < KEEP24) ? vals[e] * INV_KEEP : 0.0f;
}

// ------------------------------------------------- ego init + out[:, 0:64]=emb
__global__ void init_kernel(const float* __restrict__ emb,
                            float* __restrict__ out,
                            float* __restrict__ ego, int N) {
    int idx = blockIdx.x * blockDim.x + threadIdx.x;
    if (idx >= N * 64) return;
    int n = idx >> 6, d = idx & 63;
    float v = emb[idx];
    ego[idx] = v;
    out[(size_t)n * 256 + d] = v;
}

// ------------------------------------------------------------------ COO SpMM
// 16 threads per edge; each handles 4 consecutive floats (float4 gather +
// 4x global_atomic_add_f32 scatter).
__global__ void spmm_kernel(const int* __restrict__ rows,
                            const int* __restrict__ cols,
                            const float* __restrict__ a_vals,
                            const float* __restrict__ ego,
                            float* __restrict__ side, int E) {
    long long gid = (long long)blockIdx.x * blockDim.x + threadIdx.x;
    int e  = (int)(gid >> 4);
    int c4 = ((int)gid & 15) * 4;
    if (e >= E) return;
    float a = a_vals[e];
    if (a == 0.0f) return;
    int r = rows[e], c = cols[e];
    const float4 v = *(const float4*)(ego + (size_t)c * 64 + c4);
    float* dst = side + (size_t)r * 64 + c4;
    atomicAdd(dst + 0, a * v.x);
    atomicAdd(dst + 1, a * v.y);
    atomicAdd(dst + 2, a * v.z);
    atomicAdd(dst + 3, a * v.w);
}

// ----------------------------------------- dense layer: h = side@W + b, drop,
// write h to ego (next layer input) and h/max(||h||,eps) to out slab.
// 128 threads = 4 waves; each wave computes a 16x64 stripe via 8x
// v_wmma_f32_16x16x32_f16 (4 N-tiles x 2 K-slices).
__global__ void __launch_bounds__(128)
gemm_wmma_kernel(const float* __restrict__ side,
                 const float* __restrict__ W,
                 const float* __restrict__ bias,
                 float* __restrict__ ego,
                 float* __restrict__ out,     // pre-offset to this layer's cols
                 unsigned salt, int N) {
    __shared__ __align__(16) float    Wstage[64 * 64];   // TDM destination (f32)
    // B fragments, fragment-major: Wf[(kb2*4+nt)*512 + lane*16 + j]
    __shared__ __align__(16) _Float16 Wf[8 * 32 * 16];
    __shared__ float bl[64];
    __shared__ float hst[4][16 * 65];   // padded stride 65: no bank conflicts
    __shared__ float invn[4][16];

    const int tid = threadIdx.x;

    // ---- stage W (16 KB) into LDS via the Tensor Data Mover (wave 0 issues)
#if __has_builtin(__builtin_amdgcn_tensor_load_to_lds)
    if (tid < 32) {
        unsigned long long ga = (unsigned long long)(uintptr_t)W;
        unsigned ldsOff = (unsigned)(uintptr_t)&Wstage[0];
        // D# group0: count=1 | lds_addr | global_addr[56:0] | type=2
        u32x4 g0 = {1u, ldsOff, (unsigned)(ga & 0xFFFFFFFFu),
                    (unsigned)((ga >> 32) & 0x1FFFFFFu) | (2u << 30)};
        // D# group1: data_size=4B; tensor 64x64; tile 64x64; dim0 stride 64
        i32x8 g1 = {(int)(2u << 16),          // data_size=2 (4 bytes)
                    (int)(64u << 16),         // tensor_dim0[15:0]=64
                    (int)(64u << 16),         // tensor_dim0 hi=0 | tensor_dim1 lo=64
                    (int)(64u << 16),         // tensor_dim1 hi=0 | tile_dim0=64
                    (int)64,                  // tile_dim1=64, tile_dim2=0
                    (int)64,                  // tensor_dim0_stride[31:0]=64
                    0, 0};
        i32x4 g2 = {0, 0, 0, 0};
        i32x4 g3 = {0, 0, 0, 0};
#if __clang_major__ >= 23
        i32x8 g4 = {0, 0, 0, 0, 0, 0, 0, 0};
        __builtin_amdgcn_tensor_load_to_lds(g0, g1, g2, g3, g4, 0);
#else
        __builtin_amdgcn_tensor_load_to_lds(g0, g1, g2, g3, 0);
#endif
        __builtin_amdgcn_s_wait_tensorcnt(0);
    }
#else
    for (int i = tid; i < 64 * 64; i += 128) Wstage[i] = W[i];
#endif
    if (tid < 64) bl[tid] = bias[tid];
    __syncthreads();

    // ---- convert + swizzle W into WMMA B-fragment layout (f16).
    // Fragment (kb2,nt), lane L: b[j] = W[kb2*32 + (L<16?0:8) + (j&7) + (j>>3)*16][nt*16 + (L&15)]
    for (int idx = tid; idx < 64 * 64; idx += 128) {
        int k = idx >> 6, n = idx & 63;
        int kb2 = k >> 5, krem = k & 31;
        int hi = (krem >> 3) & 1, j = (krem & 7) | ((krem >> 4) << 3);
        int nt = n >> 4;
        int laneL = (n & 15) | (hi << 4);
        Wf[((kb2 * 4 + nt) * 32 + laneL) * 16 + j] = (_Float16)Wstage[idx];
    }
    __syncthreads();

    const int wid  = tid >> 5;
    const int lane = tid & 31;
    const int rowBase = blockIdx.x * 64 + wid * 16;

    // ---- A fragments (ISA 16-bit A 16x32 layout):
    // lane<16 : row=lane,    K = kb+{0..7}, kb+{16..23}
    // lane>=16: row=lane-16, K = kb+{8..15}, kb+{24..31}
    const int r    = lane & 15;
    const int ksel = (lane < 16) ? 0 : 8;
    const int arow = rowBase + r;
    v16h afrag[2];
#pragma unroll
    for (int kb2 = 0; kb2 < 2; ++kb2) {
        v16h a = {};
        if (arow < N) {
            const float* srow = side + (size_t)arow * 64 + kb2 * 32 + ksel;
            float4 p0 = *(const float4*)(srow + 0);
            float4 p1 = *(const float4*)(srow + 4);
            float4 p2 = *(const float4*)(srow + 16);
            float4 p3 = *(const float4*)(srow + 20);
            a[0]  = (_Float16)p0.x; a[1]  = (_Float16)p0.y;
            a[2]  = (_Float16)p0.z; a[3]  = (_Float16)p0.w;
            a[4]  = (_Float16)p1.x; a[5]  = (_Float16)p1.y;
            a[6]  = (_Float16)p1.z; a[7]  = (_Float16)p1.w;
            a[8]  = (_Float16)p2.x; a[9]  = (_Float16)p2.y;
            a[10] = (_Float16)p2.z; a[11] = (_Float16)p2.w;
            a[12] = (_Float16)p3.x; a[13] = (_Float16)p3.y;
            a[14] = (_Float16)p3.z; a[15] = (_Float16)p3.w;
        }
        afrag[kb2] = a;
    }

    // ---- 4 N-tiles; each B fragment is one contiguous 32B LDS read
    const int colL = lane & 15;
#pragma unroll
    for (int nt = 0; nt < 4; ++nt) {
        const int cb = nt * 16;
        v8f c = {};
#pragma unroll
        for (int kb2 = 0; kb2 < 2; ++kb2) {
            v16h b = *(const v16h*)&Wf[((kb2 * 4 + nt) * 32 + lane) * 16];
            c = __builtin_amdgcn_wmma_f32_16x16x32_f16(
                    false, afrag[kb2], false, b, (short)0, c, false, false);
        }
        // bias + message dropout, stage stripe in LDS.
        // C/D layout: VGPR v -> local row m = v + (lane<16 ? 0 : 8), col = lane&15.
        const int cg = cb + colL;
        const int n0 = rowBase + ((lane < 16) ? 0 : 8);
        unsigned id = (unsigned)(n0 * 64 + cg) ^ salt;
        unsigned h1 = hash_u32(id);
        unsigned h2 = hash_u32(id ^ 0x6A09E667u);
#pragma unroll
        for (int v = 0; v < 8; ++v) {
            const int m = v + ((lane < 16) ? 0 : 8);
            float h = c[v] + bl[cg];
            unsigned byte = ((v < 4 ? h1 : h2) >> (8 * (v & 3))) & 0xFFu;
            h = (byte < KEEP8) ? h * INV_KEEP : 0.0f;
            hst[wid][m * 65 + cg] = h;
        }
    }
    __syncthreads();

    // ---- per-row L2 norm: lane pair (x, x+16) splits the 64 cols of row x&15
    {
        const int rr = lane & 15;
        const int h0 = (lane < 16) ? 0 : 32;
        float s = 0.0f;
#pragma unroll
        for (int i = 0; i < 32; ++i) {
            float x = hst[wid][rr * 65 + h0 + i];
            s += x * x;
        }
        s += __shfl_xor(s, 16, 32);
        if (lane < 16) invn[wid][rr] = 1.0f / fmaxf(sqrtf(s), 1e-12f);
    }
    __syncthreads();

    // ---- write h -> ego, h*inv -> out
    for (int rr = 0; rr < 16; ++rr) {
        const int n = rowBase + rr;
        if (n >= N) break;
        const float inv = invn[wid][rr];
        for (int cc = lane; cc < 64; cc += 32) {
            float h = hst[wid][rr * 65 + cc];
            ego[(size_t)n * 64 + cc] = h;
            out[(size_t)n * 256 + cc] = h * inv;
        }
    }
}

// ---------------------------------------------------------------------- host
extern "C" void kernel_launch(void* const* d_in, const int* in_sizes, int n_in,
                              void* d_out, int out_size, void* d_ws, size_t ws_size,
                              hipStream_t stream) {
    const int*   rows = (const int*)d_in[0];
    const int*   cols = (const int*)d_in[1];
    const float* vals = (const float*)d_in[2];
    const float* emb  = (const float*)d_in[3];
    const float* Wm[3] = {(const float*)d_in[4], (const float*)d_in[6], (const float*)d_in[8]};
    const float* Bm[3] = {(const float*)d_in[5], (const float*)d_in[7], (const float*)d_in[9]};

    const int E = in_sizes[0];
    const int N = in_sizes[3] / 64;

    // workspace: a_vals[E] | ego[N*64] | side[N*64]  (~56 MB for N=100k, E=1.2M)
    float* ws     = (float*)d_ws;
    float* a_vals = ws;
    float* ego    = ws + E;
    float* side   = ego + (size_t)N * 64;
    float* out    = (float*)d_out;

    edge_drop_kernel<<<(E + 255) / 256, 256, 0, stream>>>(vals, a_vals, E);
    init_kernel<<<(N * 64 + 255) / 256, 256, 0, stream>>>(emb, out, ego, N);

    const unsigned salts[3] = {0x243F6A89u, 0x85A308D3u, 0x13198A2Eu};
    for (int l = 0; l < 3; ++l) {
        hipMemsetAsync(side, 0, (size_t)N * 64 * sizeof(float), stream);
        long long spmm_threads = (long long)E * 16;
        spmm_kernel<<<(int)((spmm_threads + 255) / 256), 256, 0, stream>>>(
            rows, cols, a_vals, ego, side, E);
        gemm_wmma_kernel<<<(N + 63) / 64, 128, 0, stream>>>(
            side, Wm[l], Bm[l], ego, out + 64 * (l + 1), salts[l], N);
    }
}
